// DMSK_26285199851632
// MI455X (gfx1250) — matrix operand order repaired and checked by reading, not compiled, not run on hardware
//
#include <hip/hip_runtime.h>
#include <math.h>

// ---------------------------------------------------------------------------
// Problem constants (from reference)
// ---------------------------------------------------------------------------
#define BATCH 16
#define CHN   128          // C
#define CH2   64           // C/2
#define HH    160
#define WW2   160
#define PX    (HH * WW2)   // 25600 pixels per channel plane

typedef __attribute__((ext_vector_type(16))) _Float16 v16h;
typedef __attribute__((ext_vector_type(8)))  float    v8f;

// ---------------------------------------------------------------------------
// Kernel 1: channel projection (1x1 conv C->C2) via WMMA f16->f32.
//   out[b, 64+m, p] = sum_c proj_w[m,c] * x[b,c,p]
// x_proj is staged in the UPPER 64 channels of d_out (overwritten later by att2).
// Each wave: one 16-pixel N-tile, full M=64 (4 tiles), K=128 in 4 chunks of 32.
// ---------------------------------------------------------------------------
__global__ __launch_bounds__(256) void dmsk_proj_wmma(
    const float* __restrict__ x, const float* __restrict__ proj_w,
    float* __restrict__ out)
{
  __shared__ _Float16 wsh[CH2 * CHN];           // 16 KB: proj_w in f16
  for (int i = threadIdx.x; i < CH2 * CHN; i += 256)
    wsh[i] = (_Float16)proj_w[i];
  __syncthreads();

  const int lane = threadIdx.x & 31;
  const int wave = threadIdx.x >> 5;            // 0..7
  const int n    = lane & 15;                   // column (pixel) within tile
  const int h    = lane >> 4;                   // lane half
  const int b    = blockIdx.y;
  const int p0   = blockIdx.x * 128 + wave * 16;

  const float* xb = x + (size_t)b * CHN * PX;

  v8f acc0 = {0,0,0,0,0,0,0,0};
  v8f acc1 = {0,0,0,0,0,0,0,0};
  v8f acc2 = {0,0,0,0,0,0,0,0};
  v8f acc3 = {0,0,0,0,0,0,0,0};

  for (int k0 = 0; k0 < CHN; k0 += 32) {
    // B fragment (x tile, 32x16 K-major): element j -> K = k0 + h*16 + j, N = n
    v16h bf;
#pragma unroll
    for (int j = 0; j < 16; ++j)
      bf[j] = (_Float16)xb[(size_t)(k0 + h * 16 + j) * PX + p0 + n];

    // A fragments (weights 16x32): element j -> K = k0 + h*16 + j
#pragma unroll
    for (int t = 0; t < 4; ++t) {
      const int m = t * 16 + n;                 // A-matrix row = lane%16
      v16h af;
#pragma unroll
      for (int j = 0; j < 16; ++j) {
        const int kk = k0 + h * 16 + j;
        af[j] = wsh[m * CHN + kk];
      }
      v8f c = (t == 0) ? acc0 : (t == 1) ? acc1 : (t == 2) ? acc2 : acc3;
      c = __builtin_amdgcn_wmma_f32_16x16x32_f16(false, af, false, bf,
                                                 (short)0, c, false, false);
      if (t == 0) acc0 = c; else if (t == 1) acc1 = c;
      else if (t == 2) acc2 = c; else acc3 = c;
    }
  }

  // D layout: VGPR r -> M = t*16 + h*8 + r, N = p0 + n. Write to channels 64..127.
  float* ob = out + (size_t)b * CHN * PX + (size_t)CH2 * PX;
#pragma unroll
  for (int t = 0; t < 4; ++t) {
    const v8f c = (t == 0) ? acc0 : (t == 1) ? acc1 : (t == 2) ? acc2 : acc3;
#pragma unroll
    for (int r = 0; r < 8; ++r)
      ob[(size_t)(t * 16 + h * 8 + r) * PX + p0 + n] = c[r];
  }
}

// ---------------------------------------------------------------------------
// Kernel 2: global average pool over pixels for one (b, c) plane.
// base layout: [B][CHN][PX]; pools channels [chan_off, chan_off+64).
// ---------------------------------------------------------------------------
__global__ __launch_bounds__(256) void dmsk_pool(
    const float* __restrict__ base, int chan_off, float* __restrict__ pooled)
{
  __shared__ float red[256];
  const int c = blockIdx.x, b = blockIdx.y, tid = threadIdx.x;
  const float* sp = base + (size_t)b * CHN * PX + (size_t)(chan_off + c) * PX;
  float s = 0.f;
  for (int p = tid; p < PX; p += 256) s += sp[p];
  red[tid] = s;
  __syncthreads();
  for (int st = 128; st > 0; st >>= 1) {
    if (tid < st) red[tid] += red[tid + st];
    __syncthreads();
  }
  if (tid == 0) pooled[b * CH2 + c] = red[0] / (float)PX;
}

// ---------------------------------------------------------------------------
// Kernel 3: routing — logits = pooled @ att_w.T + att_b, argmax (first max wins,
// matching jnp.argmax; softmax is monotone so it can be skipped).
// ---------------------------------------------------------------------------
__global__ void dmsk_route(const float* __restrict__ pooled,
                           const float* __restrict__ att_w,
                           const float* __restrict__ att_b,
                           int nexp, int* __restrict__ idx)
{
  const int b = threadIdx.x;
  if (b >= BATCH) return;
  const float* p = pooled + b * CH2;
  float best = -INFINITY; int bi = 0;
  for (int e = 0; e < nexp; ++e) {
    float s = att_b[e];
    for (int c = 0; c < CH2; ++c) s += att_w[e * CH2 + c] * p[c];
    if (s > best) { best = s; bi = e; }
  }
  idx[b] = bi;
}

// ---------------------------------------------------------------------------
// Kernel 4: depthwise conv with per-sample expert (kernel size) selection.
// src channels [src_off, src_off+64), dst channels [dst_off, dst_off+64).
// ---------------------------------------------------------------------------
__global__ __launch_bounds__(256) void dmsk_dwconv(
    const float* __restrict__ src, int src_off,
    float* __restrict__ dst, int dst_off,
    const float* __restrict__ w0, const float* __restrict__ b0, int k0,
    const float* __restrict__ w1, const float* __restrict__ b1, int k1,
    const float* __restrict__ w2, const float* __restrict__ b2, int k2,
    const int* __restrict__ idx, int dil)
{
  const int p = blockIdx.x * 256 + threadIdx.x;      // pixel
  const int c = blockIdx.y, b = blockIdx.z;
  const int hh = p / WW2, ww = p % WW2;

  const int e = idx[b];
  const float* wgt; const float* bias; int k;
  if (e == 0)      { wgt = w0; bias = b0; k = k0; }
  else if (e == 1) { wgt = w1; bias = b1; k = k1; }
  else             { wgt = w2; bias = b2; k = k2; }
  const int kh = k >> 1;

  const float* sp = src + (size_t)b * CHN * PX + (size_t)(src_off + c) * PX;
  const float* wc = wgt + c * k * k;
  float acc = bias[c];
  for (int i = 0; i < k; ++i) {
    const int y = hh + (i - kh) * dil;
    if ((unsigned)y >= (unsigned)HH) continue;
    const float* row = sp + y * WW2;
    __builtin_prefetch(row + ww, 0, 1);              // global_prefetch_b8
    for (int j = 0; j < k; ++j) {
      const int xx = ww + (j - kh) * dil;
      if ((unsigned)xx >= (unsigned)WW2) continue;
      acc += wc[i * k + j] * row[xx];
    }
  }
  dst[(size_t)b * CHN * PX + (size_t)(dst_off + c) * PX + p] = acc;
}

// ---------------------------------------------------------------------------
// Kernel 5: channel-wise avg & max maps -> amap[b][2][PX]
// ---------------------------------------------------------------------------
__global__ __launch_bounds__(256) void dmsk_avgmax(
    const float* __restrict__ out, float* __restrict__ amap)
{
  const int p = blockIdx.x * 256 + threadIdx.x;
  const int b = blockIdx.y;
  const float* ob = out + (size_t)b * CHN * PX + p;
  float s = 0.f, mx = -INFINITY;
  for (int c = 0; c < CHN; ++c) {
    const float v = ob[(size_t)c * PX];
    s += v; mx = fmaxf(mx, v);
  }
  amap[((size_t)b * 2 + 0) * PX + p] = s / (float)CHN;
  amap[((size_t)b * 2 + 1) * PX + p] = mx;
}

// ---------------------------------------------------------------------------
// Kernel 6: 7x7 spatial-attention conv (2->2), sigmoid gate, in-place
//           out = out * (sig(a0) + sig(a1)) + x
// ---------------------------------------------------------------------------
__global__ __launch_bounds__(256) void dmsk_gate(
    const float* __restrict__ amap, const float* __restrict__ sp_w,
    const float* __restrict__ sp_b, const float* __restrict__ x,
    float* __restrict__ out)
{
  const int p = blockIdx.x * 256 + threadIdx.x;
  const int b = blockIdx.y;
  const int hh = p / WW2, ww = p % WW2;

  float z0 = sp_b[0], z1 = sp_b[1];
  for (int i = 0; i < 2; ++i) {
    const float* am = amap + ((size_t)b * 2 + i) * PX;
    const float* w0 = sp_w + (0 * 2 + i) * 49;
    const float* w1 = sp_w + (1 * 2 + i) * 49;
    for (int r = 0; r < 7; ++r) {
      const int y = hh + r - 3;
      if ((unsigned)y >= (unsigned)HH) continue;
      const float* row = am + y * WW2;
      for (int s = 0; s < 7; ++s) {
        const int xx = ww + s - 3;
        if ((unsigned)xx >= (unsigned)WW2) continue;
        const float v = row[xx];
        z0 += w0[r * 7 + s] * v;
        z1 += w1[r * 7 + s] * v;
      }
    }
  }
  const float g = 1.f / (1.f + expf(-z0)) + 1.f / (1.f + expf(-z1));

  float* ob = out + (size_t)b * CHN * PX + p;
  const float* xb = x + (size_t)b * CHN * PX + p;
  for (int c = 0; c < CHN; ++c)
    ob[(size_t)c * PX] = ob[(size_t)c * PX] * g + xb[(size_t)c * PX];
}

// ---------------------------------------------------------------------------
// Launch: inputs in setup_inputs() order.
// ---------------------------------------------------------------------------
extern "C" void kernel_launch(void* const* d_in, const int* in_sizes, int n_in,
                              void* d_out, int out_size, void* d_ws, size_t ws_size,
                              hipStream_t stream) {
  const float* x       = (const float*)d_in[0];
  const float* proj_w  = (const float*)d_in[1];
  const float* dw1_w3  = (const float*)d_in[2];
  const float* dw1_b3  = (const float*)d_in[3];
  const float* dw1_w5  = (const float*)d_in[4];
  const float* dw1_b5  = (const float*)d_in[5];
  const float* dw2_w7  = (const float*)d_in[6];
  const float* dw2_b7  = (const float*)d_in[7];
  const float* dw2_w9  = (const float*)d_in[8];
  const float* dw2_b9  = (const float*)d_in[9];
  const float* dw2_w11 = (const float*)d_in[10];
  const float* dw2_b11 = (const float*)d_in[11];
  const float* att1_w  = (const float*)d_in[12];
  const float* att1_b  = (const float*)d_in[13];
  const float* att2_w  = (const float*)d_in[14];
  const float* att2_b  = (const float*)d_in[15];
  const float* sp_w    = (const float*)d_in[16];
  const float* sp_b    = (const float*)d_in[17];
  float* out = (float*)d_out;

  // Workspace: pooled (B*C2) | amap (B*2*PX) | idx (B ints)  ~= 3.3 MB
  float* pooled = (float*)d_ws;
  float* amap   = pooled + BATCH * CH2;
  int*   idx    = (int*)(amap + (size_t)BATCH * 2 * PX);

  // 1) x_proj -> upper 64 channels of d_out (WMMA GEMM)
  dmsk_proj_wmma<<<dim3(PX / 128, BATCH), 256, 0, stream>>>(x, proj_w, out);

  // 2) route 1: pool x_proj (channels 64..127), argmax over 2 experts
  dmsk_pool<<<dim3(CH2, BATCH), 256, 0, stream>>>(out, CH2, pooled);
  dmsk_route<<<1, 32, 0, stream>>>(pooled, att1_w, att1_b, 2, idx);

  // 3) att1 = selected dwconv(x_proj), k in {3,5}, dil 1 -> channels 0..63
  dmsk_dwconv<<<dim3(PX / 256, CH2, BATCH), 256, 0, stream>>>(
      out, CH2, out, 0,
      dw1_w3, dw1_b3, 3, dw1_w5, dw1_b5, 5, dw1_w5, dw1_b5, 5, idx, 1);

  // 4) route 2: pool att1 (channels 0..63), argmax over 3 experts
  dmsk_pool<<<dim3(CH2, BATCH), 256, 0, stream>>>(out, 0, pooled);
  dmsk_route<<<1, 32, 0, stream>>>(pooled, att2_w, att2_b, 3, idx);

  // 5) att2 = selected dwconv(att1), k in {7,9,11}, dil 3 -> channels 64..127
  dmsk_dwconv<<<dim3(PX / 256, CH2, BATCH), 256, 0, stream>>>(
      out, 0, out, CH2,
      dw2_w7, dw2_b7, 7, dw2_w9, dw2_b9, 9, dw2_w11, dw2_b11, 11, idx, 3);

  // 6) channel avg/max maps
  dmsk_avgmax<<<dim3(PX / 256, BATCH), 256, 0, stream>>>(out, amap);

  // 7) 7x7 spatial conv + sigmoid gate + residual, in place
  dmsk_gate<<<dim3(PX / 256, BATCH), 256, 0, stream>>>(amap, sp_w, sp_b, x, out);
}
